// QuantMLP_69483980914906
// MI455X (gfx1250) — 8-bit matrix operand order repaired and checked by reading, compiled, not run on hardware
//
#include <hip/hip_runtime.h>
#include <hip/hip_bf16.h>
#include <stdint.h>


typedef __attribute__((ext_vector_type(8))) int v8i;
typedef __attribute__((ext_vector_type(2))) int v2i;

#define DI __device__ __forceinline__

constexpr int BM   = 128;
constexpr int BN   = 128;
constexpr int BK   = 64;
constexpr int LDSK = 80;

DI int q8(float v, float qs) {
  float f = fminf(fmaxf(v * qs, -127.0f), 127.0f);
  return __float2int_rn(f);
}

DI void async_ld128(void* lds, const void* gptr) {
  unsigned int       l = (unsigned int)(uintptr_t)lds;
  unsigned long long g = (unsigned long long)(uintptr_t)gptr;
  asm volatile("global_load_async_to_lds_b128 %0, %1, off" :: "v"(l), "v"(g) : "memory");
}
DI void wait_async0() { asm volatile("s_wait_asynccnt 0x0" ::: "memory"); }

__global__ void quantize_x_kernel(const float* __restrict__ x,
                                  const float* __restrict__ a_in,
                                  int8_t* __restrict__ q, int n4) {
  int i = blockIdx.x * blockDim.x + threadIdx.x;
  if (i >= n4) return;
  const float qs = 127.0f / a_in[0];
  float4 v = ((const float4*)x)[i];
  int p = (q8(v.x, qs) & 255) | ((q8(v.y, qs) & 255) << 8) |
          ((q8(v.z, qs) & 255) << 16) | ((q8(v.w, qs) & 255) << 24);
  ((int*)q)[i] = p;
}

__global__ void pack_w_kernel(const int* __restrict__ W, int8_t* __restrict__ Wq,
                              int* __restrict__ bias, int K) {
  const int n = blockIdx.y;
  const int k = blockIdx.x * blockDim.x + threadIdx.x;
  const int* row = W + (size_t)n * (K + 1);
  if (k < K)       Wq[(size_t)n * K + k] = (int8_t)row[k];
  else if (k == K) bias[n] = row[K];
}

template <bool QUANT_OUT>
__global__ __launch_bounds__(256)
void qgemm_kernel(const int8_t* __restrict__ A,
                  const int8_t* __restrict__ W,
                  const int*    __restrict__ bias,
                  int K, int N,
                  const float* __restrict__ a_in_p,
                  const float* __restrict__ a_w_p,
                  const float* __restrict__ a_next_p,
                  int8_t* __restrict__ qout,
                  float*  __restrict__ fout) {
  __shared__ __align__(16) int8_t sA[2][BM * LDSK];
  __shared__ __align__(16) int8_t sW[2][BN * LDSK];

  const int tid = threadIdx.x;
  const int bm  = blockIdx.y * BM;
  const int bn  = blockIdx.x * BN;

  const int c0  = tid, c1 = tid + 256;
  const int ar0 = c0 >> 2, ac0 = (c0 & 3) * 16;
  const int ar1 = c1 >> 2, ac1 = (c1 & 3) * 16;

  const int8_t* gA = A + (size_t)bm * K;
  const int8_t* gW = W + (size_t)bn * K;
  const int nk = K / BK;

  async_ld128(&sA[0][ar0 * LDSK + ac0], gA + (size_t)ar0 * K + ac0);
  async_ld128(&sA[0][ar1 * LDSK + ac1], gA + (size_t)ar1 * K + ac1);
  async_ld128(&sW[0][ar0 * LDSK + ac0], gW + (size_t)ar0 * K + ac0);
  async_ld128(&sW[0][ar1 * LDSK + ac1], gW + (size_t)ar1 * K + ac1);
  wait_async0();
  __syncthreads();

  const int w    = tid >> 5;
  const int lane = tid & 31;
  const int h    = lane >> 4;
  const int r    = lane & 15;
  const int wm   = (w & 3) * 32;
  const int wn   = (w >> 2) * 64;

  v8i acc[2][4] = {};

  for (int ks = 0; ks < nk; ++ks) {
    const int  buf  = ks & 1;
    const bool more = (ks + 1) < nk;

    if (more) {
      const size_t koff = (size_t)(ks + 1) * BK;
      const int nb = buf ^ 1;
      async_ld128(&sA[nb][ar0 * LDSK + ac0], gA + (size_t)ar0 * K + koff + ac0);
      async_ld128(&sA[nb][ar1 * LDSK + ac1], gA + (size_t)ar1 * K + koff + ac1);
      async_ld128(&sW[nb][ar0 * LDSK + ac0], gW + (size_t)ar0 * K + koff + ac0);
      async_ld128(&sW[nb][ar1 * LDSK + ac1], gW + (size_t)ar1 * K + koff + ac1);
      if (ks + 2 < nk) {
        __builtin_prefetch(gA + (size_t)ar0 * K + koff + BK + ac0, 0, 0);
        __builtin_prefetch(gW + (size_t)ar0 * K + koff + BK + ac0, 0, 0);
      }
    }

    v8i af[2], bf[4];
#pragma unroll
    for (int i = 0; i < 2; ++i) {
      const int8_t* base = &sA[buf][(wm + i * 16 + r) * LDSK + h * 8];
#pragma unroll
      for (int p = 0; p < 4; ++p) {
        v2i d = *(const v2i*)(base + (p >> 1) * 32 + (p & 1) * 16);
        af[i][2 * p] = d.x; af[i][2 * p + 1] = d.y;
      }
    }
#pragma unroll
    for (int j = 0; j < 4; ++j) {
      const int8_t* base = &sW[buf][(wn + j * 16 + r) * LDSK + h * 8];
#pragma unroll
      for (int p = 0; p < 4; ++p) {
        v2i d = *(const v2i*)(base + (p >> 1) * 32 + (p & 1) * 16);
        bf[j][2 * p] = d.x; bf[j][2 * p + 1] = d.y;
      }
    }

#pragma unroll
    for (int i = 0; i < 2; ++i)
#pragma unroll
      for (int j = 0; j < 4; ++j)
        acc[i][j] = __builtin_amdgcn_wmma_i32_16x16x64_iu8(
            true, af[i], true, bf[j], acc[i][j], false, false);

    if (more) wait_async0();
    __syncthreads();
  }

  const float s = a_in_p[0] * a_w_p[0] * (1.0f / 16129.0f);
  float qs = 0.0f;
  if (QUANT_OUT) qs = 127.0f / a_next_p[0];

#pragma unroll
  for (int j = 0; j < 4; ++j) {
    const int col = bn + wn + j * 16 + r;
    const int bc  = bias[col];
#pragma unroll
    for (int i = 0; i < 2; ++i) {
      const int row0 = bm + wm + i * 16 + h * 8;
#pragma unroll
      for (int v = 0; v < 8; ++v) {
        const int accv = acc[i][j][v] + bc;
        float f = (float)accv * s;
        if (QUANT_OUT) {
          f = fmaxf(f, 0.0f);
          qout[(size_t)(row0 + v) * N + col] = (int8_t)q8(f, qs);
        } else {
          fout[(size_t)(row0 + v) * N + col] = f;
        }
      }
    }
  }
}

extern "C" void kernel_launch(void* const* d_in, const int* in_sizes, int n_in,
                              void* d_out, int out_size, void* d_ws, size_t ws_size,
                              hipStream_t stream) {
  (void)in_sizes; (void)n_in; (void)out_size; (void)ws_size;
  const float* x     = (const float*)d_in[0];
  const float* a_in0 = (const float*)d_in[1];
  const float* a_w0  = (const float*)d_in[2];
  const float* a_in2 = (const float*)d_in[3];
  const float* a_w2  = (const float*)d_in[4];
  const float* a_in4 = (const float*)d_in[5];
  const float* a_w4  = (const float*)d_in[6];
  const int*   W0    = (const int*)d_in[7];
  const int*   W1    = (const int*)d_in[8];
  const int*   W2    = (const int*)d_in[9];
  float* out = (float*)d_out;

  char* ws = (char*)d_ws;
  int8_t* q0  = (int8_t*)(ws + 0);
  int8_t* q1  = (int8_t*)(ws + (size_t)(4u  << 20));
  int8_t* q2  = (int8_t*)(ws + (size_t)(20u << 20));
  int8_t* W0q = (int8_t*)(ws + (size_t)(36u << 20));
  int8_t* W1q = (int8_t*)(ws + (size_t)(40u << 20));
  int8_t* W2q = (int8_t*)(ws + (size_t)(56u << 20));
  int*    b0  = (int*)(ws + (size_t)(60u << 20));
  int*    b1  = (int*)(ws + (size_t)(60u << 20) + 65536);
  int*    b2  = (int*)(ws + (size_t)(60u << 20) + 131072);

  const int n4 = (4096 * 1024) / 4;
  quantize_x_kernel<<<(n4 + 255) / 256, 256, 0, stream>>>(x, a_in0, q0, n4);

  pack_w_kernel<<<dim3((1025 + 255) / 256, 4096), 256, 0, stream>>>(W0, W0q, b0, 1024);
  pack_w_kernel<<<dim3((4097 + 255) / 256, 4096), 256, 0, stream>>>(W1, W1q, b1, 4096);
  pack_w_kernel<<<dim3((4097 + 255) / 256, 1024), 256, 0, stream>>>(W2, W2q, b2, 4096);

  qgemm_kernel<true ><<<dim3(4096 / BN, 4096 / BM), 256, 0, stream>>>(
      q0, W0q, b0, 1024, 4096, a_in0, a_w0, a_in2, q1, nullptr);
  qgemm_kernel<true ><<<dim3(4096 / BN, 4096 / BM), 256, 0, stream>>>(
      q1, W1q, b1, 4096, 4096, a_in2, a_w2, a_in4, q2, nullptr);
  qgemm_kernel<false><<<dim3(1024 / BN, 4096 / BM), 256, 0, stream>>>(
      q2, W2q, b2, 4096, 1024, a_in4, a_w4, nullptr, nullptr, out);
}
